// Tagger_65489661329564
// MI455X (gfx1250) — compile-verified, matrix operand order checked by reading, not run on hardware
//
#include <hip/hip_runtime.h>
#include <stdint.h>

typedef __attribute__((ext_vector_type(16))) _Float16 v16h;
typedef __attribute__((ext_vector_type(8)))  float    v8f;
typedef __attribute__((ext_vector_type(4)))  float    v4f;

#define Bn 1024
#define Tn 256
#define Nn 64

#define WAVES_PER_BLOCK 4
#define BATCH_PER_WAVE 16
#define NBLOCKS (Bn / (WAVES_PER_BLOCK * BATCH_PER_WAVE))   // 16
#define TOTAL_WAVES (Bn / BATCH_PER_WAVE)                   // 64

#define LOG2E 1.44269504088896340736f
#define LN2   0.69314718055994530942f

// K-index pattern for the 16-bit A/B WMMA fragment:
// VGPR r (0..7), dword half hi (0/1), lane half h (0/1)
__device__ __forceinline__ int kof(int r, int hi, int h) {
  return ((r & 4) ? 16 : 0) + h * 8 + 2 * (r & 3) + hi;
}

// raw hardware transcendentals (base-2); inputs here are always normal-range
__device__ __forceinline__ float rlog2(float x) { return __builtin_amdgcn_logf(x); }
__device__ __forceinline__ float rexp2(float x) { return __builtin_amdgcn_exp2f(x); }

// ---------------------------------------------------------------------------
// Prep: normalize transitions in log space; precompute
//   wsMcol2[k]   = log2e * max_j trans_n[j][k]
//   Eexp[j][k]   = exp(trans_n[j][k] - M_k)            (f16, linear domain)
//   wsStrans[k]  = strans_n[k]                          (natural log)
//   wsEtrans2[k] = log2e * etrans_n[k]                  (log2 domain)
// One block of 64 threads (2 waves).
// ---------------------------------------------------------------------------
__global__ void crf_prep(const float* __restrict__ trans,
                         const float* __restrict__ strans,
                         const float* __restrict__ etrans,
                         float* __restrict__ wsMcol2,
                         float* __restrict__ wsStrans,
                         float* __restrict__ wsEtrans2,
                         _Float16* __restrict__ Eexp) {
  __shared__ float tn[64][65];
  const int j = threadIdx.x;  // 0..63

  // row-wise logsumexp of trans row j
  float m = -3.4e38f;
  for (int k = 0; k < 64; ++k) m = fmaxf(m, trans[j * 64 + k]);
  float s = 0.f;
  for (int k = 0; k < 64; ++k) s += __expf(trans[j * 64 + k] - m);
  const float rl = m + __logf(s);
  for (int k = 0; k < 64; ++k) tn[j][k] = trans[j * 64 + k] - rl;
  __syncthreads();

  // column max + exp-normalized f16 matrix (thread j handles column k=j)
  const int k = j;
  float cm = -3.4e38f;
  for (int jj = 0; jj < 64; ++jj) cm = fmaxf(cm, tn[jj][k]);
  wsMcol2[k] = LOG2E * cm;
  for (int jj = 0; jj < 64; ++jj)
    Eexp[jj * 64 + k] = (_Float16)__expf(tn[jj][k] - cm);

  // strans / etrans log-normalization (computed redundantly per thread)
  float ms = -3.4e38f;
  for (int i = 0; i < 64; ++i) ms = fmaxf(ms, strans[i]);
  float ss = 0.f;
  for (int i = 0; i < 64; ++i) ss += __expf(strans[i] - ms);
  wsStrans[j] = strans[j] - (ms + __logf(ss));

  float me = -3.4e38f;
  for (int i = 0; i < 64; ++i) me = fmaxf(me, etrans[i]);
  float se = 0.f;
  for (int i = 0; i < 64; ++i) se += __expf(etrans[i] - me);
  wsEtrans2[j] = LOG2E * (etrans[j] - (me + __logf(se)));
}

// ---------------------------------------------------------------------------
// Main scan kernel. One wave32 per 16 batches. Alpha is kept in the LOG2
// domain (alpha2 = log2e * alpha) so the scan uses the raw base-2 hardware
// transcendentals. A = E^T lives in registers (8 v16h fragments); B =
// exp2(alpha2 - m2_b) rebuilt each step from LDS. D accumulators: lane =
// batch, VGPR index = output label. Mask word + emit b128 loads issued at
// the step top so exp/WMMA work covers their latency.
// ---------------------------------------------------------------------------
__global__ void __launch_bounds__(WAVES_PER_BLOCK * 32)
crf_forward(const float* __restrict__ emit,
            const uint8_t* __restrict__ mask,
            const float* __restrict__ wsMcol2,
            const float* __restrict__ wsStrans,
            const float* __restrict__ wsEtrans2,
            const _Float16* __restrict__ Eexp,
            float* __restrict__ partials) {
  __shared__ float aSh[WAVES_PER_BLOCK][BATCH_PER_WAVE][Nn + 4];

  const int lane = threadIdx.x & 31;
  const int wv   = threadIdx.x >> 5;
  const int h    = lane >> 4;   // lane half
  const int bl   = lane & 15;   // batch within wave tile
  const int waveGlobal = blockIdx.x * WAVES_PER_BLOCK + wv;
  const int wb = waveGlobal * BATCH_PER_WAVE;
  const int bg = wb + bl;       // this lane's global batch

  float (*al)[Nn + 4] = aSh[wv];

  // ---- constant A fragments: A[mt][kc](m_local, j_local) = E[j][mt*16+m] ----
  v16h A[4][2];
#pragma unroll
  for (int mt = 0; mt < 4; ++mt)
#pragma unroll
    for (int kc = 0; kc < 2; ++kc)
#pragma unroll
      for (int r = 0; r < 8; ++r)
#pragma unroll
        for (int hi = 0; hi < 2; ++hi) {
          const int j  = kc * 32 + kof(r, hi, h);
          const int ko = mt * 16 + bl;
          A[mt][kc][2 * r + hi] = Eexp[j * Nn + ko];
        }

  // ---- per-lane column maxes (log2 domain) in D layout k = mt*16+v+8h ----
  float Mc[32];
#pragma unroll
  for (int mt = 0; mt < 4; ++mt)
#pragma unroll
    for (int v = 0; v < 8; ++v)
      Mc[mt * 8 + v] = wsMcol2[mt * 16 + v + 8 * h];

  // ---- alpha2_0 = log2e * (strans_n + emit[:,0,:]) ----
  const float* eb = emit + (size_t)bg * Tn * Nn;
#pragma unroll
  for (int mt = 0; mt < 4; ++mt)
#pragma unroll
    for (int v = 0; v < 8; ++v) {
      const int k = mt * 16 + v + 8 * h;
      al[bl][k] = LOG2E * (wsStrans[k] + eb[k]);
    }
  __syncthreads();

  // mask row for this lane's batch, read as aligned 32-bit words
  const uint32_t* mrow = (const uint32_t*)(mask + (size_t)bg * Tn);
  uint32_t mw = 0;

  // ---- sequential scan over time ----
  for (int t = 1; t < Tn; ++t) {
    // --- issue this step's memory ops first; consume ~40 instrs later ---
    if ((t & 3) == 0 || t == 1) mw = mrow[t >> 2];  // 1 word / 4 steps

    const float* pe = eb + (size_t)t * Nn;
    v4f ev[8];  // this lane's 32 emit values as 8x b128 loads
#pragma unroll
    for (int mt = 0; mt < 4; ++mt) {
      const v4f* p = (const v4f*)(pe + mt * 16 + 8 * h);
      ev[2 * mt]     = p[0];
      ev[2 * mt + 1] = p[1];
    }
    if (t + 1 < Tn) __builtin_prefetch(pe + Nn, 0, 1);  // global_prefetch_b8

    // --- build B fragments: exp2(alpha2[b][j] - m2_b), f16, WMMA B layout ---
    float raw[32];
    float mx = -3.4e38f;
#pragma unroll
    for (int kc = 0; kc < 2; ++kc)
#pragma unroll
      for (int r = 0; r < 8; ++r)
#pragma unroll
        for (int hi = 0; hi < 2; ++hi) {
          const float x = al[bl][kc * 32 + kof(r, hi, h)];
          raw[kc * 16 + 2 * r + hi] = x;
          mx = fmaxf(mx, x);
        }
    // lanes l and l^16 share a batch and together see all 64 labels
    mx = fmaxf(mx, __shfl_xor(mx, 16, 32));

    v16h Bf[2];
#pragma unroll
    for (int kc = 0; kc < 2; ++kc)
#pragma unroll
      for (int i = 0; i < 16; ++i)
        Bf[kc][i] = (_Float16)rexp2(raw[kc * 16 + i] - mx);

    // --- D[k_out, b] = sum_j E[j,k_out] * expAlpha[b,j] (linear domain) ---
    v8f acc[4];
#pragma unroll
    for (int mt = 0; mt < 4; ++mt) {
      v8f z = {};
      acc[mt] = __builtin_amdgcn_wmma_f32_16x16x32_f16(
          false, A[mt][0], false, Bf[0], (short)0, z, false, false);
      acc[mt] = __builtin_amdgcn_wmma_f32_16x16x32_f16(
          false, A[mt][1], false, Bf[1], (short)0, acc[mt], false, false);
    }

    // --- masked update: alpha2 = m2 + M2_k + log2(D) + log2e*emit ---
    const bool live = ((mw >> ((t & 3) * 8)) & 0xffu) != 0u;
    if (live) {
#pragma unroll
      for (int mt = 0; mt < 4; ++mt)
#pragma unroll
        for (int v = 0; v < 8; ++v) {
          const int k = mt * 16 + v + 8 * h;
          const float e = ev[2 * mt + (v >> 2)][v & 3];
          al[bl][k] = fmaf(e, LOG2E, mx + Mc[mt * 8 + v]) + rlog2(acc[mt][v]);
        }
    }
    __syncthreads();  // order LDS writeback vs next step's cross-lane reads
  }

  // ---- logZ contribution: ln2 * sum_b lse2_k(alpha2[b,:] + etrans2) ----
  float part = 0.0f;
  if (lane < 16) {
    float vals[64];
    float mm = -3.4e38f;
#pragma unroll
    for (int k = 0; k < 64; ++k) {
      const float x = al[lane][k] + wsEtrans2[k];
      vals[k] = x;
      mm = fmaxf(mm, x);
    }
    float s = 0.f;
#pragma unroll
    for (int k = 0; k < 64; ++k) s += rexp2(vals[k] - mm);
    part = LN2 * (mm + rlog2(s));
  }
#pragma unroll
  for (int off = 16; off >= 1; off >>= 1)
    part += __shfl_xor(part, off, 32);
  if (lane == 0) partials[waveGlobal] = part;
}

// ---------------------------------------------------------------------------
// Deterministic final reduction (fixed summation order, no float atomics).
// ---------------------------------------------------------------------------
__global__ void crf_final(const float* __restrict__ partials,
                          float* __restrict__ out) {
  if (threadIdx.x == 0 && blockIdx.x == 0) {
    float s = 0.f;
    for (int i = 0; i < TOTAL_WAVES; ++i) s += partials[i];
    out[0] = s;
  }
}

extern "C" void kernel_launch(void* const* d_in, const int* in_sizes, int n_in,
                              void* d_out, int out_size, void* d_ws, size_t ws_size,
                              hipStream_t stream) {
  const float*   emit   = (const float*)d_in[0];
  const float*   trans  = (const float*)d_in[1];
  const float*   strans = (const float*)d_in[2];
  const float*   etrans = (const float*)d_in[3];
  const uint8_t* mask   = (const uint8_t*)d_in[4];

  float* ws        = (float*)d_ws;
  float* wsMcol2   = ws;                 // 64 f32 (log2 domain)
  float* wsStrans  = ws + 64;            // 64 f32 (natural log)
  float* wsEtrans2 = ws + 128;           // 64 f32 (log2 domain)
  _Float16* Eexp   = (_Float16*)(ws + 192);   // 64*64 f16 (= 2048 f32 slots)
  float* partials  = ws + 192 + 2048;    // 64 f32

  crf_prep<<<1, 64, 0, stream>>>(trans, strans, etrans,
                                 wsMcol2, wsStrans, wsEtrans2, Eexp);
  crf_forward<<<NBLOCKS, WAVES_PER_BLOCK * 32, 0, stream>>>(
      emit, mask, wsMcol2, wsStrans, wsEtrans2, Eexp, partials);
  crf_final<<<1, 32, 0, stream>>>(partials, (float*)d_out);
}
